// LinearAttention_60327110639895
// MI455X (gfx1250) — matrix operand order repaired and checked by reading, not compiled, not run on hardware
//
#include <hip/hip_runtime.h>

typedef __bf16 bf16;
typedef __attribute__((ext_vector_type(16))) __bf16 v16bf;
typedef __attribute__((ext_vector_type(8)))  __bf16 v8bf;
typedef __attribute__((ext_vector_type(8)))  float  v8f;

#define N_TOK   16384   // B*S
#define DM      1024
#define S_LEN   8192
#define NH      16
#define DK      64
#define KV_SPLIT 16
#define KV_COLS  80     // 64 V cols + 1 ones col (K-sum) + 15 zero pad -> 5 n-tiles

#define PADROW_F 1028   // f32 LDS row stride: 1028 dwords -> rows 4 banks apart
#define PADROW_B 1032   // bf16 LDS row stride: 516 dwords -> rows 4 banks apart

// ---------------------------------------------------------------------------
// WMMA helper (CDNA5 wave32, 16x16x32 bf16 -> f32)
// ---------------------------------------------------------------------------
__device__ __forceinline__ v8f wmma_bf16(v16bf a, v16bf b, v8f c) {
  return __builtin_amdgcn_wmma_f32_16x16x32_bf16(false, a, false, b, (short)0, c,
                                                 false, false);
}

// A fragment (16x32, 16-bit): lane L holds row M=L%16.
// half=L/16: VGPR0-3 = K = half*8 + [0..7], VGPR4-7 = K = 16 + half*8 + [0..7]
__device__ __forceinline__ v16bf ldA_bf16(const bf16* row, int kk, int half) {
  union { v16bf v; v8bf h[2]; } u;
  u.h[0] = *(const v8bf*)(row + kk + half * 8);
  u.h[1] = *(const v8bf*)(row + kk + 16 + half * 8);
  return u.v;
}
// B fragment (32x16, 16-bit): lane L holds col N=L%16; K = half*8 + [0..7], 16 + half*8 + [0..7]
__device__ __forceinline__ v16bf ldB_bf16(const bf16* row, int kk, int half) {
  union { v16bf v; v8bf h[2]; } u;
  u.h[0] = *(const v8bf*)(row + kk + half * 8);
  u.h[1] = *(const v8bf*)(row + kk + 16 + half * 8);
  return u.v;
}

// Async global->LDS copy of one 16-byte chunk (CDNA5 async-memory path,
// tracked by ASYNCcnt).  lds_addr is the low-32 LDS byte address.
__device__ __forceinline__ void async_cp_b128(unsigned lds_addr,
                                              const void* gptr) {
  asm volatile("global_load_async_to_lds_b128 %0, %1, off"
               :: "v"(lds_addr), "v"((unsigned long long)gptr)
               : "memory");
}
__device__ __forceinline__ void wait_async_all() {
  asm volatile("s_wait_asynccnt 0" ::: "memory");
}

// ---------------------------------------------------------------------------
// fp32 -> bf16 convert (weights)
// ---------------------------------------------------------------------------
__global__ void cvt_bf16_kernel(const float* __restrict__ in,
                                bf16* __restrict__ out, int n) {
  int i = blockIdx.x * 256 + threadIdx.x;
  if (i < n) out[i] = (bf16)in[i];
}

// ---------------------------------------------------------------------------
// Projection GEMM: C(N_TOK x DM) = act(X @ W^T + b).
// One block = one 16-token strip; 16 waves = the 16 output blocks of 64.
// The X strip is staged into LDS once via async-to-LDS, then every wave
// builds its A fragments from LDS (16x reduction in global A traffic).
// ACT: 0=none, 1=elu+1 feature map
// OUTMODE: 0 = bf16 scatter to (b,h,s,d)   [Q]
//          1 = bf16 scatter to (b,h,d,s)   [K^T, V^T]
//          2 = fp32 row-major (n, DM)      [final output]
// ---------------------------------------------------------------------------
template <int ACT, int OUTMODE, bool A_BF16>
__global__ void proj_kernel(const float* __restrict__ Xf,
                            const bf16* __restrict__ Xb,
                            const bf16* __restrict__ W,
                            const float* __restrict__ bias,
                            void* __restrict__ out) {
  __shared__ __align__(16) char smem[16 * PADROW_F * 4];
  float* tf  = (float*)smem;
  bf16*  tbf = (bf16*)smem;

  const int tid  = threadIdx.x;
  const int lane = tid & 31;
  const int tb   = blockIdx.x;       // token block (16 tokens)
  const int ob   = tid >> 5;         // wave index == output block of 64
  const int col = lane & 15, half = lane >> 4;

  // ---- stage X strip (16 rows x 1024) into LDS with async b128 copies ----
  if (A_BF16) {
    // 16 rows x 2048B = 2048 chunks of 16B; 4 per thread
    for (int c = tid; c < 2048; c += 512) {
      const int row = c >> 7, cc = c & 127;
      const bf16* g = Xb + ((size_t)(tb * 16 + row)) * DM + cc * 8;
      unsigned la = (unsigned)(unsigned long long)(tbf + row * PADROW_B + cc * 8);
      async_cp_b128(la, g);
    }
  } else {
    // 16 rows x 4096B = 4096 chunks of 16B; 8 per thread
    for (int c = tid; c < 4096; c += 512) {
      const int row = c >> 8, cc = c & 255;
      const float* g = Xf + ((size_t)(tb * 16 + row)) * DM + cc * 4;
      unsigned la = (unsigned)(unsigned long long)(tf + row * PADROW_F + cc * 4);
      async_cp_b128(la, g);
    }
  }
  wait_async_all();   // own wave's ASYNCcnt == 0
  __syncthreads();    // all waves' staging visible

  const bf16* wrow[4];
#pragma unroll
  for (int t = 0; t < 4; ++t)
    wrow[t] = W + (size_t)(ob * 64 + t * 16 + col) * DM;

  v8f acc[4] = {v8f{}, v8f{}, v8f{}, v8f{}};
  for (int kk = 0; kk < DM; kk += 32) {
    // A fragment from LDS (row = lane&15), converting f32->bf16 if needed
    v16bf a;
    if (A_BF16) {
      const bf16* r0 = tbf + col * PADROW_B;
      union { v16bf v; v8bf h[2]; } u;
      u.h[0] = *(const v8bf*)(r0 + kk + half * 8);
      u.h[1] = *(const v8bf*)(r0 + kk + 16 + half * 8);
      a = u.v;
    } else {
      const float* p0 = tf + col * PADROW_F + kk + half * 8;
      const float* p1 = tf + col * PADROW_F + kk + 16 + half * 8;
#pragma unroll
      for (int i = 0; i < 8; ++i) { a[i] = (bf16)p0[i]; a[8 + i] = (bf16)p1[i]; }
    }
#pragma unroll
    for (int t = 0; t < 4; ++t)
      acc[t] = wmma_bf16(a, ldB_bf16(wrow[t], kk, half), acc[t]);
  }

#pragma unroll
  for (int t = 0; t < 4; ++t) {
    const int o = ob * 64 + t * 16 + col;    // output feature (C column)
    const float bv = bias[o];
#pragma unroll
    for (int r = 0; r < 8; ++r) {
      float v = acc[t][r] + bv;
      if (ACT == 1) v = (v > 0.0f) ? (v + 1.0f) : __expf(v);  // elu(x)+1
      const int n = tb * 16 + r + half * 8;  // token (C row)
      if (OUTMODE == 2) {
        ((float*)out)[(size_t)n * DM + o] = v;
      } else {
        const int b = n >> 13, s = n & (S_LEN - 1);
        const int h = o >> 6, d = o & (DK - 1);
        const size_t idx =
            (OUTMODE == 0) ? (((size_t)(b * NH + h) * S_LEN + s) * DK + d)
                           : (((size_t)(b * NH + h) * DK + d) * S_LEN + s);
        ((bf16*)out)[idx] = (bf16)v;
      }
    }
  }
}

// ---------------------------------------------------------------------------
// KV stage: per (b,h): KV_aug(64 x 80) = K^T @ [V | 1 | 0pad] over S chunk.
// Kt/Vt are (b,h, d, s) bf16 so the reduction dim s is contiguous.
// grid (32 heads, KV_SPLIT), block 128 (4 waves; wave = one 16-row d block)
// ---------------------------------------------------------------------------
__global__ void kv_kernel(const bf16* __restrict__ Kt,
                          const bf16* __restrict__ Vt,
                          float* __restrict__ part) {
  const int tid = threadIdx.x, lane = tid & 31, w = tid >> 5;
  const int bh = blockIdx.x, split = blockIdx.y;
  const int col = lane & 15, half = lane >> 4;

  const bf16* krow = Kt + ((size_t)bh * DK + w * 16 + col) * S_LEN;
  const bf16* vrow[4];
#pragma unroll
  for (int t = 0; t < 4; ++t)
    vrow[t] = Vt + ((size_t)bh * DK + t * 16 + col) * S_LEN;

  // ones column (col 0 of 5th tile) folds K-sum into the same WMMA pass
  v16bf ones;
#pragma unroll
  for (int i = 0; i < 16; ++i) ones[i] = (bf16)((col == 0) ? 1.0f : 0.0f);

  v8f acc[5] = {v8f{}, v8f{}, v8f{}, v8f{}, v8f{}};
  const int s0 = split * (S_LEN / KV_SPLIT);
  for (int s = s0; s < s0 + (S_LEN / KV_SPLIT); s += 32) {
    v16bf a = ldA_bf16(krow, s, half);
#pragma unroll
    for (int t = 0; t < 4; ++t)
      acc[t] = wmma_bf16(a, ldB_bf16(vrow[t], s, half), acc[t]);
    acc[4] = wmma_bf16(a, ones, acc[4]);
  }

  float* p = part + ((size_t)bh * KV_SPLIT + split) * (DK * KV_COLS);
#pragma unroll
  for (int t = 0; t < 5; ++t)
#pragma unroll
    for (int r = 0; r < 8; ++r) {
      const int d = w * 16 + r + half * 8;
      const int e = t * 16 + col;
      p[d * KV_COLS + e] = acc[t][r];
    }
}

// Deterministic fixed-order reduction of partials; emit transposed bf16
// KVT[bh][e][d] so d is the contiguous reduction dim for the Q@KV GEMM.
__global__ void kv_reduce_kernel(const float* __restrict__ part,
                                 bf16* __restrict__ KVT) {
  const int idx = blockIdx.x * 256 + threadIdx.x;   // 32 * 64 * 80
  if (idx >= 32 * DK * KV_COLS) return;
  const int bh = idx / (DK * KV_COLS);
  const int rem = idx % (DK * KV_COLS);
  const int d = rem / KV_COLS, e = rem % KV_COLS;
  float s = 0.0f;
  for (int sp = 0; sp < KV_SPLIT; ++sp)
    s += part[((size_t)bh * KV_SPLIT + sp) * (DK * KV_COLS) + rem];
  KVT[(size_t)bh * (KV_COLS * DK) + e * DK + d] = (bf16)s;
}

// ---------------------------------------------------------------------------
// out = (Q @ KV_aug) / (Q . Ksum + eps); wave = 16 tokens of one (b,h).
// Column 64 of the augmented result IS the normalizer -> broadcast across
// row lanes with ds_bpermute (wave32 lane&16 trick).
// ---------------------------------------------------------------------------
__global__ void attn_kernel(const bf16* __restrict__ Qh,
                            const bf16* __restrict__ KVT,
                            bf16* __restrict__ attn) {
  const int tid = threadIdx.x, lane = tid & 31;
  const int wid = blockIdx.x * 8 + (tid >> 5);     // 16384 waves
  const int sb = wid & 511, h = (wid >> 9) & 15, b = wid >> 13;
  const int col = lane & 15, half = lane >> 4;
  const int bh = b * NH + h;

  const bf16* qrow = Qh + ((size_t)bh * S_LEN + sb * 16 + col) * DK;
  const bf16* kvrow[5];
#pragma unroll
  for (int t = 0; t < 5; ++t)
    kvrow[t] = KVT + (size_t)bh * (KV_COLS * DK) + (t * 16 + col) * DK;

  v8f acc[5] = {v8f{}, v8f{}, v8f{}, v8f{}, v8f{}};
#pragma unroll
  for (int kk = 0; kk < DK; kk += 32) {
    v16bf a = ldA_bf16(qrow, kk, half);
#pragma unroll
    for (int t = 0; t < 5; ++t)
      acc[t] = wmma_bf16(a, ldB_bf16(kvrow[t], kk, half), acc[t]);
  }

  // normalizer lives in lanes 0 / 16 of tile 4 (global col 64)
  const int bidx = (lane & 16) * 4;  // byte index for ds_bpermute
#pragma unroll
  for (int r = 0; r < 8; ++r) {
    const float nv = __int_as_float(
        __builtin_amdgcn_ds_bpermute(bidx, __float_as_int(acc[4][r])));
    const float inv = 1.0f / (nv + 1e-6f);
#pragma unroll
    for (int t = 0; t < 4; ++t) acc[t][r] *= inv;
  }

#pragma unroll
  for (int t = 0; t < 4; ++t)
#pragma unroll
    for (int r = 0; r < 8; ++r) {
      const int s = sb * 16 + r + half * 8;
      const int o = h * DK + t * 16 + col;
      attn[((size_t)b * S_LEN + s) * DM + o] = (bf16)acc[t][r];
    }
}

// ---------------------------------------------------------------------------
extern "C" void kernel_launch(void* const* d_in, const int* in_sizes, int n_in,
                              void* d_out, int out_size, void* d_ws, size_t ws_size,
                              hipStream_t stream) {
  const float* q_in = (const float*)d_in[0];
  const float* k_in = (const float*)d_in[1];
  const float* v_in = (const float*)d_in[2];
  const float* W_q = (const float*)d_in[3];
  const float* b_q = (const float*)d_in[4];
  const float* W_k = (const float*)d_in[5];
  const float* b_k = (const float*)d_in[6];
  const float* W_v = (const float*)d_in[7];
  const float* b_v = (const float*)d_in[8];
  const float* W_o = (const float*)d_in[9];
  const float* b_o = (const float*)d_in[10];

  char* ws = (char*)d_ws;
  const size_t WBYTES = (size_t)DM * DM * sizeof(bf16);        // 2 MB
  bf16* Wq_b = (bf16*)(ws + 0 * WBYTES);
  bf16* Wk_b = (bf16*)(ws + 1 * WBYTES);
  bf16* Wv_b = (bf16*)(ws + 2 * WBYTES);
  bf16* Wo_b = (bf16*)(ws + 3 * WBYTES);
  size_t off = 4 * WBYTES;
  bf16* Qh   = (bf16*)(ws + off); off += (size_t)N_TOK * DM * sizeof(bf16);   // 32 MB
  bf16* Kt   = (bf16*)(ws + off); off += (size_t)N_TOK * DM * sizeof(bf16);   // 32 MB
  bf16* Vt   = (bf16*)(ws + off); off += (size_t)N_TOK * DM * sizeof(bf16);   // 32 MB
  bf16* attn = (bf16*)(ws + off); off += (size_t)N_TOK * DM * sizeof(bf16);   // 32 MB
  float* part = (float*)(ws + off);
  off += (size_t)32 * KV_SPLIT * DK * KV_COLS * sizeof(float);                // 10 MB
  bf16* KVT = (bf16*)(ws + off);

  // 1) weights -> bf16
  const int wN = DM * DM;
  cvt_bf16_kernel<<<(wN + 255) / 256, 256, 0, stream>>>(W_q, Wq_b, wN);
  cvt_bf16_kernel<<<(wN + 255) / 256, 256, 0, stream>>>(W_k, Wk_b, wN);
  cvt_bf16_kernel<<<(wN + 255) / 256, 256, 0, stream>>>(W_v, Wv_b, wN);
  cvt_bf16_kernel<<<(wN + 255) / 256, 256, 0, stream>>>(W_o, Wo_b, wN);

  // 2) projections: 1024 blocks, 16 waves each (one token strip per block)
  proj_kernel<1, 0, false><<<1024, 512, 0, stream>>>(q_in, nullptr, Wq_b, b_q, Qh);
  proj_kernel<1, 1, false><<<1024, 512, 0, stream>>>(k_in, nullptr, Wk_b, b_k, Kt);
  proj_kernel<0, 1, false><<<1024, 512, 0, stream>>>(v_in, nullptr, Wv_b, b_v, Vt);

  // 3) KV aggregation (deterministic split + fixed-order reduce)
  kv_kernel<<<dim3(32, KV_SPLIT), 128, 0, stream>>>(Kt, Vt, part);
  kv_reduce_kernel<<<(32 * DK * KV_COLS + 255) / 256, 256, 0, stream>>>(part, KVT);

  // 4) Q @ KV_aug + normalize
  attn_kernel<<<2048, 256, 0, stream>>>(Qh, KVT, attn);

  // 5) output projection -> fp32 d_out
  proj_kernel<0, 2, true><<<1024, 512, 0, stream>>>(nullptr, attn, Wo_b, b_o,
                                                    (float*)d_out);
}
